// ConditionalGNN_11553462026720
// MI455X (gfx1250) — matrix operand order repaired and checked by reading, not compiled, not run on hardware
//
#include <hip/hip_runtime.h>

// ---------------------------------------------------------------------------
// ConditionalGNN for MI455X (gfx1250): bf16 WMMA GEMMs + atomic edge scatter
// ---------------------------------------------------------------------------

typedef __bf16 bf16_t;
typedef bf16_t v16bf __attribute__((ext_vector_type(16)));
typedef float  v8f   __attribute__((ext_vector_type(8)));

#define HID 128
#define KSLAB 128
#define LDSW 136   // 128 + 8 pad (16B) to break LDS bank conflicts

__device__ __forceinline__ bf16_t f2bf(float f) {
  union { float f; unsigned u; } x; x.f = f;
  unsigned r = x.u + 0x7FFFu + ((x.u >> 16) & 1u);   // round-to-nearest-even
  unsigned short hbits = (unsigned short)(r >> 16);
  return __builtin_bit_cast(bf16_t, hbits);
}

__device__ __forceinline__ void atomAddF(float* p, float v) {
  unsafeAtomicAdd(p, v);   // lowers to global_atomic_add_f32
}

// ---------------------------------------------------------------------------
// GEMM: out[N x 128] = Abf16 @ Wbf16 with f32 accumulation.
//   cond != nullptr : A row = concat(x[node, 0:768], cond[batch[node], 0:768]), K=1536
//   cond == nullptr : A row = A[node, 0:K]                                     K=128
// One block = 16 nodes x 128 outputs. 8 waves; wave w owns N-tile [16w,16w+16).
// K is staged through LDS in 128-wide slabs, W transposed to [n][k] in LDS.
// ---------------------------------------------------------------------------
__global__ __launch_bounds__(256) void gemm_node_bf16(
    const float* __restrict__ A, const float* __restrict__ cond,
    const int* __restrict__ batch, const float* __restrict__ W,
    float* __restrict__ out, int N, int K)
{
  __shared__ __align__(16) bf16_t ldsA[16 * LDSW];
  __shared__ __align__(16) bf16_t ldsB[HID * LDSW];

  const int tid    = threadIdx.x;
  const int m0     = blockIdx.x * 16;
  const int wave   = tid >> 5;
  const int lane   = tid & 31;
  const int lane16 = lane & 15;
  const int hi     = lane >> 4;       // 0: lanes 0-15, 1: lanes 16-31
  const int n0     = wave * 16;

  v8f c = {};  // f32 accumulator, 16x16 tile

  for (int ks = 0; ks < K; ks += KSLAB) {
    // --- stage A slab: 16 rows x 128 K (f32 -> bf16) ---
    for (int i = tid; i < 16 * KSLAB; i += 256) {
      int r = i >> 7, k = i & (KSLAB - 1);
      int node = m0 + r;
      int kk = ks + k;
      float f = 0.0f;
      if (node < N) {
        if (cond) f = (kk < 768) ? A[(size_t)node * 768 + kk]
                                 : cond[(size_t)batch[node] * 768 + (kk - 768)];
        else      f = A[(size_t)node * K + kk];
      }
      ldsA[r * LDSW + k] = f2bf(f);
    }
    // --- stage W slab transposed: ldsB[n][k] = W[ks+k][n] ---
    for (int i = tid; i < HID * KSLAB; i += 256) {
      int k = i >> 7, n = i & (HID - 1);
      ldsB[n * LDSW + k] = f2bf(W[(size_t)(ks + k) * HID + n]);
    }
    __syncthreads();

#pragma unroll
    for (int ki = 0; ki < KSLAB / 32; ++ki) {
      const int kb = ki * 32;
      union { v16bf v; uint4 q[2]; } a, b;
      // A 16x32 bf16 fragment: row = lane16; lanes<16: K {0..7,16..23}, lanes>=16: +8
      const bf16_t* pa = &ldsA[lane16 * LDSW + kb + hi * 8];
      a.q[0] = *(const uint4*)(pa);
      a.q[1] = *(const uint4*)(pa + 16);
      // B 32x16 bf16 fragment: col = lane16; lanes<16: K {0..7,16..23}, lanes>=16: +8
      const bf16_t* pb = &ldsB[(n0 + lane16) * LDSW + kb + hi * 8];
      b.q[0] = *(const uint4*)(pb);
      b.q[1] = *(const uint4*)(pb + 16);
      c = __builtin_amdgcn_wmma_f32_16x16x32_bf16(false, a.v, false, b.v,
                                                  (short)0, c, false, false);
    }
    __syncthreads();
  }

  // C/D layout: VGPR v -> M = v + 8*hi, N = lane16
#pragma unroll
  for (int v = 0; v < 8; ++v) {
    int row = m0 + v + hi * 8;
    if (row < N) out[(size_t)row * HID + n0 + lane16] = c[v];
  }
}

// ---------------------------------------------------------------------------
// Graph normalization + message passing
// ---------------------------------------------------------------------------
__global__ void init_deg(float* deg, int N) {
  int i = blockIdx.x * blockDim.x + threadIdx.x;
  if (i < N) deg[i] = 1.0f;             // self-loop contribution
}

__global__ void deg_edges(const int* __restrict__ dst, float* deg, int E) {
  int e = blockIdx.x * blockDim.x + threadIdx.x;
  if (e < E) atomAddF(&deg[dst[e]], 1.0f);
}

__global__ void finish_dinv(float* deg, int N) {
  int i = blockIdx.x * blockDim.x + threadIdx.x;
  if (i < N) deg[i] = rsqrtf(deg[i]);   // deg >= 1 always (self-loops)
}

// acc[node][c] = t[node][c] * dinv[node]^2   (self-loop message, also inits acc)
__global__ void self_msg(const float* __restrict__ t, const float* __restrict__ dinv,
                         float* __restrict__ acc, int total) {
  int i = blockIdx.x * blockDim.x + threadIdx.x;
  if (i < total) {
    float di = dinv[i >> 7];
    acc[i] = t[i] * di * di;
  }
}

// one wave per edge; lane handles 4 channels (32*4 = 128)
__global__ void edge_msg(const int* __restrict__ src, const int* __restrict__ dst,
                         const float* __restrict__ dinv, const float* __restrict__ t,
                         float* __restrict__ acc, int E) {
  int gid  = blockIdx.x * blockDim.x + threadIdx.x;
  int e    = gid >> 5;
  int lane = gid & 31;
  if (e >= E) return;
  int s = src[e], d = dst[e];
  float nm = dinv[s] * dinv[d];
  float4 m = ((const float4*)(t + (size_t)s * HID))[lane];
  float* p = acc + (size_t)d * HID + lane * 4;
  atomAddF(p + 0, m.x * nm);
  atomAddF(p + 1, m.y * nm);
  atomAddF(p + 2, m.z * nm);
  atomAddF(p + 3, m.w * nm);
}

__global__ void relu_bias(const float* __restrict__ acc, const float* __restrict__ b,
                          float* __restrict__ h, int total) {
  int i = blockIdx.x * blockDim.x + threadIdx.x;
  if (i < total) {
    float v = acc[i] + b[i & (HID - 1)];
    h[i] = v > 0.0f ? v : 0.0f;
  }
}

// logits[node] = dot(h[node,:], Wout) + bout ; one wave per node
__global__ void out_proj(const float* __restrict__ h, const float* __restrict__ Wout,
                         const float* __restrict__ bout, float* __restrict__ out, int N) {
  int gid  = blockIdx.x * blockDim.x + threadIdx.x;
  int node = gid >> 5;
  int lane = gid & 31;
  if (node >= N) return;
  float4 hv = ((const float4*)(h + (size_t)node * HID))[lane];
  float4 wv = ((const float4*)Wout)[lane];
  float p = hv.x * wv.x + hv.y * wv.y + hv.z * wv.z + hv.w * wv.w;
#pragma unroll
  for (int off = 16; off > 0; off >>= 1) p += __shfl_down(p, off, 32);
  if (lane == 0) out[node] = p + bout[0];
}

// ---------------------------------------------------------------------------
extern "C" void kernel_launch(void* const* d_in, const int* in_sizes, int n_in,
                              void* d_out, int out_size, void* d_ws, size_t ws_size,
                              hipStream_t stream) {
  const float* x     = (const float*)d_in[0];
  const int*   ei    = (const int*)  d_in[1];
  const float* cond  = (const float*)d_in[2];
  const int*   batch = (const int*)  d_in[3];
  const float* W1    = (const float*)d_in[4];
  const float* b1    = (const float*)d_in[5];
  const float* W2    = (const float*)d_in[6];
  const float* b2    = (const float*)d_in[7];
  const float* Wout  = (const float*)d_in[8];
  const float* bout  = (const float*)d_in[9];
  float* out = (float*)d_out;

  const int N  = in_sizes[0] / 768;     // 50000
  const int E  = in_sizes[1] / 2;       // 600000
  const int K1 = in_sizes[4] / HID;     // 1536
  const int* src = ei;
  const int* dst = ei + E;

  // workspace: dinv | t | acc | h   (all 256B aligned)
  char* ws = (char*)d_ws;
  size_t o = 0;
  float* dinv = (float*)(ws + o); o += ((size_t)N * 4 + 255) & ~(size_t)255;
  float* t    = (float*)(ws + o); o += ((size_t)N * HID * 4 + 255) & ~(size_t)255;
  float* acc  = (float*)(ws + o); o += ((size_t)N * HID * 4 + 255) & ~(size_t)255;
  float* h    = (float*)(ws + o);

  const int total = N * HID;
  const int gemmBlocks = (N + 15) / 16;

  // normalization
  init_deg   <<<(N + 255) / 256, 256, 0, stream>>>(dinv, N);
  deg_edges  <<<(E + 255) / 256, 256, 0, stream>>>(dst, dinv, E);
  finish_dinv<<<(N + 255) / 256, 256, 0, stream>>>(dinv, N);

  // layer 1: t = concat(x, cond[batch]) @ W1 ; propagate ; relu(+b1)
  gemm_node_bf16<<<gemmBlocks, 256, 0, stream>>>(x, cond, batch, W1, t, N, K1);
  self_msg <<<(total + 255) / 256, 256, 0, stream>>>(t, dinv, acc, total);
  edge_msg <<<(E * 32 + 255) / 256, 256, 0, stream>>>(src, dst, dinv, t, acc, E);
  relu_bias<<<(total + 255) / 256, 256, 0, stream>>>(acc, b1, h, total);

  // layer 2: t = h @ W2 ; propagate ; relu(+b2)
  gemm_node_bf16<<<gemmBlocks, 256, 0, stream>>>(h, nullptr, nullptr, W2, t, N, HID);
  self_msg <<<(total + 255) / 256, 256, 0, stream>>>(t, dinv, acc, total);
  edge_msg <<<(E * 32 + 255) / 256, 256, 0, stream>>>(src, dst, dinv, t, acc, E);
  relu_bias<<<(total + 255) / 256, 256, 0, stream>>>(acc, b2, h, total);

  // output projection
  out_proj<<<(N * 32 + 255) / 256, 256, 0, stream>>>(h, Wout, bout, out, N);
}
